// MultiHead_14542759264515
// MI455X (gfx1250) — matrix operand order repaired and checked by reading, not compiled, not run on hardware
//
#include <hip/hip_runtime.h>

// ---------------------------------------------------------------------------
// Axial multi-head attention (b=8,c=64,h=w=128, NH=4, d_k=16) for gfx1250.
// 4096 independent 16x128 attention problems per branch; one block handles
// both branches of one problem and writes the summed LeakyReLU output.
// Matmuls: v_wmma_f32_16x16x32_f16 (fp32 accumulate). Softmax fully in
// registers via wave shuffles; all LDS WMMA operand accesses are b128.
// ---------------------------------------------------------------------------

typedef __attribute__((ext_vector_type(16))) _Float16 v16h;
typedef __attribute__((ext_vector_type(8)))  _Float16 v8h;
typedef __attribute__((ext_vector_type(4)))  _Float16 v4h;
typedef __attribute__((ext_vector_type(8)))  float    v8f;

#define SEQ    128
#define CDIM   64
#define SCALE  0.25f   // 1/sqrt(d_k), d_k = 16
#define SLOPE  0.2f

__device__ __forceinline__ v8f wmma16(v16h a, v16h b, v8f c) {
  // (neg_a, A, neg_b, B, c_mod, C, reuse_a, reuse_b)
  return __builtin_amdgcn_wmma_f32_16x16x32_f16(false, a, false, b, (short)0, c,
                                                false, false);
}

__device__ __forceinline__ v16h cat8(v8h lo, v8h hi) {
  return __builtin_shufflevector(lo, hi, 0, 1, 2, 3, 4, 5, 6, 7,
                                 8, 9, 10, 11, 12, 13, 14, 15);
}

// A-operand lane layout: lanes 0-15 hold K {0..15}, lanes 16-31 hold
// K {16..31}.  Element e sits at K = base + hh*16 + e: 16 contiguous halves
// (two 16 B chunks).
__device__ __forceinline__ v16h load_a_chunks(const _Float16* p /*16B aligned*/) {
  v8h lo = *(const v8h*)(p);
  v8h hi = *(const v8h*)(p + 8);
  return cat8(lo, hi);
}

// Project a 32x64 f16 tile by W^T (out[i][j] = sum_k in[i][k]*W[j][k]).
// TMAJOR=false: store flat (== d-major 16x128: addr i*64+j).
// TMAJOR=true : store t-major 128x16 (addr t*16+d) for the Q^T Q stage.
template <bool TMAJOR>
__device__ __forceinline__ void proj_tile(const _Float16* __restrict__ sIn,
                                          const _Float16* __restrict__ sWm,
                                          _Float16* __restrict__ sOut,
                                          int wave, int lane) {
  const int hh = lane >> 4, lm = lane & 15;
  const int mi = wave >> 2, nj = wave & 3;   // 2x4 tiles of 16x16
  v8f acc = {};
#pragma unroll
  for (int kk = 0; kk < 2; ++kk) {
    v16h a = load_a_chunks(sIn + (mi * 16 + lm) * CDIM + kk * 32 + hh * 16);
    // B[K=k][N=j] = W[j][k]; 16 contiguous halves (2 x b128)
    v16h b = *(const v16h*)(sWm + (nj * 16 + lm) * CDIM + kk * 32 + hh * 16);
    acc = wmma16(a, b, acc);
  }
#pragma unroll
  for (int r = 0; r < 8; ++r) {
    const int i = mi * 16 + hh * 8 + r;      // row of 32x64 result
    const int j = nj * 16 + lm;
    if (TMAJOR) {
      const int d = i >> 1, t = ((i & 1) << 6) | j;
      sOut[t * 16 + d] = (_Float16)acc[r];
    } else {
      sOut[i * CDIM + j] = (_Float16)acc[r]; // == sOut[d*128 + t]
    }
  }
}

// S = Q^T Q * scale with fused register softmax; writes attention transposed
// (sAt[t][s], f16).  Wave w owns s-tile w; its C tiles hold, per lane,
// 8 rows s = w*16+hh*8+r at a fixed t = tt*16+lm -> row-reductions are a
// register reduction over tt plus 4 shuffle-xor steps over lm.
__device__ __forceinline__ void scores_softmax(const _Float16* __restrict__ sQt,
                                               _Float16* __restrict__ sAt,
                                               int wave, int lane) {
  const int hh = lane >> 4, lm = lane & 15;

  // A[M=s][K=d]: lanes 0-15 hold K=d 0..15 (two b128); K 16..31 zero-padded.
  v8h qa = *(const v8h*)(sQt + (wave * 16 + lm) * 16);
  v8h qb = *(const v8h*)(sQt + (wave * 16 + lm) * 16 + 8);
  v16h za = {};
  v16h a = hh ? za : cat8(qa, qb);

  v8f c[8];
#pragma unroll
  for (int tt = 0; tt < 8; ++tt) {
    // B[K=d][N=t]: 16 contiguous halves; lanes 16-31 carry K>=16 -> zeros.
    v16h q = *(const v16h*)(sQt + (tt * 16 + lm) * 16);
    v16h zb = {};
    v16h b = hh ? zb : q;
    v8f cz = {};
    c[tt] = wmma16(a, b, cz);
  }

  float mx[8], sum[8];
#pragma unroll
  for (int r = 0; r < 8; ++r) mx[r] = -1e30f;
#pragma unroll
  for (int tt = 0; tt < 8; ++tt)
#pragma unroll
    for (int r = 0; r < 8; ++r) {
      c[tt][r] *= SCALE;
      mx[r] = fmaxf(mx[r], c[tt][r]);
    }
#pragma unroll
  for (int m = 1; m <= 8; m <<= 1)
#pragma unroll
    for (int r = 0; r < 8; ++r)
      mx[r] = fmaxf(mx[r], __shfl_xor(mx[r], m, 32));

#pragma unroll
  for (int r = 0; r < 8; ++r) sum[r] = 0.0f;
#pragma unroll
  for (int tt = 0; tt < 8; ++tt)
#pragma unroll
    for (int r = 0; r < 8; ++r) {
      const float e = __expf(c[tt][r] - mx[r]);
      c[tt][r] = e;
      sum[r] += e;
    }
#pragma unroll
  for (int m = 1; m <= 8; m <<= 1)
#pragma unroll
    for (int r = 0; r < 8; ++r)
      sum[r] += __shfl_xor(sum[r], m, 32);

  float inv[8];
#pragma unroll
  for (int r = 0; r < 8; ++r) inv[r] = 1.0f / sum[r];

  // attn[s][t] -> sAt[t*128 + s]; 8 consecutive s per lane = one b128 store.
#pragma unroll
  for (int tt = 0; tt < 8; ++tt) {
    v8h o;
#pragma unroll
    for (int r = 0; r < 8; ++r) o[r] = (_Float16)(c[tt][r] * inv[r]);
    *(v8h*)(sAt + (tt * 16 + lm) * SEQ + wave * 16 + hh * 8) = o;
  }
}

// ctx[d][t] = sum_s V[d][s] * A[s][t]; wave w owns t-tile w; 4 k-steps of 32.
__device__ __forceinline__ v8f attn_ctx(const _Float16* __restrict__ sV,
                                        const _Float16* __restrict__ sAt,
                                        int wave, int lane) {
  const int hh = lane >> 4, lm = lane & 15;
  v8f acc = {};
#pragma unroll
  for (int kk = 0; kk < 4; ++kk) {
    v16h a = load_a_chunks(sV + lm * SEQ + kk * 32 + hh * 16);
    // B[K=s][N=t]: 16 contiguous halves of sAt row t.
    v16h b = *(const v16h*)(sAt + (wave * 16 + lm) * SEQ + kk * 32 + hh * 16);
    acc = wmma16(a, b, acc);
  }
  return acc;
}

__device__ __forceinline__ v4h cvt4(float4 f) {
  v4h h;
  h[0] = (_Float16)f.x; h[1] = (_Float16)f.y;
  h[2] = (_Float16)f.z; h[3] = (_Float16)f.w;
  return h;
}

__global__ void __launch_bounds__(256)
axial_mha_kernel(const float* __restrict__ z, const float* __restrict__ WQ,
                 const float* __restrict__ WK, float* __restrict__ out) {
  extern __shared__ char smem[];
  _Float16* sW    = (_Float16*)(smem);          // [2][64*64] 16 KB (WQ,WK)
  _Float16* sTile = (_Float16*)(smem + 16384);  // [32][64]    4 KB staging
  _Float16* sQt   = (_Float16*)(smem + 20480);  // [128][16]   4 KB (t-major Q)
  _Float16* sV    = (_Float16*)(smem + 24576);  // [16][128]   4 KB
  _Float16* sAt   = (_Float16*)(smem + 28672);  // [128][128] 32 KB attn^T
                                                // total 60 KB

  const int tid  = threadIdx.x;
  const int wave = tid >> 5, lane = tid & 31;
  const int hh = lane >> 4, lm = lane & 15;

  const int N    = blockIdx.x;        // 0..4095 problem id
  const int n    = N >> 2;            // batch-row index (b*128 + h|w)
  const int head = N & 3;
  const int bi   = n >> 7;
  const int hw   = n & 127;
  const size_t zimg = (size_t)bi * (64 * 128 * 128);

  // ---- stage weights (f16, vectorized) ----
#pragma unroll
  for (int k = 0; k < 4; ++k) {
    const int v = tid + k * 256;                // 0..1023 float4s
    *(v4h*)(sW + v * 4)        = cvt4(((const float4*)WQ)[v]);
    *(v4h*)(sW + 4096 + v * 4) = cvt4(((const float4*)WK)[v]);
  }
  // ---- stage V raw rows (2048 contiguous floats) ----
  {
    const float4* vz = (const float4*)(z + (size_t)N * 2048);
#pragma unroll
    for (int k = 0; k < 2; ++k) {
      const int v = tid + k * 256;              // 0..511
      *(v4h*)(sTile + v * 4) = cvt4(vz[v]);
    }
  }
  __syncthreads();

  proj_tile<false>(sTile, sW + 4096, sV, wave, lane); // V = raw @ W_K^T
  __syncthreads();

  v8f ctx0, ctx1;
#pragma unroll
  for (int br = 0; br < 2; ++br) {
    // ---- gather branch-specific 32x64 z-tile feeding W_Q ----
    if (br == 0) {
      // a: z[bi, ci, hw, head*32+i] -- contiguous in i: float4 loads
      const float* base = z + zimg + (size_t)hw * 128 + head * 32;
#pragma unroll
      for (int k = 0; k < 2; ++k) {
        const int v  = tid + k * 256;           // 0..511
        const int ci = v >> 3;
        const int i4 = (v & 7) * 4;
        const float4 f = *(const float4*)(base + (size_t)ci * 16384 + i4);
        sTile[(i4 + 0) * 64 + ci] = (_Float16)f.x;
        sTile[(i4 + 1) * 64 + ci] = (_Float16)f.y;
        sTile[(i4 + 2) * 64 + ci] = (_Float16)f.z;
        sTile[(i4 + 3) * 64 + ci] = (_Float16)f.w;
      }
    } else {
      // m: z[bi, ci, head*32+i, hw] -- inherent 512 B-stride gather
      const float* base = z + zimg + (size_t)head * 32 * 128 + hw;
#pragma unroll
      for (int k = 0; k < 8; ++k) {
        const int idx = tid + k * 256;          // 0..2047
        const int ci = idx >> 5, i = idx & 31;
        sTile[i * 64 + ci] =
            (_Float16)base[(size_t)ci * 16384 + (size_t)i * 128];
      }
    }
    __syncthreads();

    proj_tile<true>(sTile, sW, sQt, wave, lane);   // Q (t-major)
    __syncthreads();

    scores_softmax(sQt, sAt, wave, lane);          // softmax(Q^T Q * scale)^T
    __syncthreads();

    const v8f c = attn_ctx(sV, sAt, wave, lane);   // ctx = V @ A
    if (br == 0) ctx0 = c; else ctx1 = c;
    __syncthreads();                               // before buffers reused
  }

  // ---- combine LeakyReLU(ctx0)+LeakyReLU(ctx1), store (coalesced) ----
#pragma unroll
  for (int r = 0; r < 8; ++r) {
    const int d = hh * 8 + r;
    const int t = wave * 16 + lm;
    const float x0 = ctx0[r], x1 = ctx1[r];
    const float y0 = (x0 > 0.0f) ? x0 : SLOPE * x0;
    const float y1 = (x1 > 0.0f) ? x1 : SLOPE * x1;
    out[(size_t)N * 2048 + d * SEQ + t] = y0 + y1;
  }
}

extern "C" void kernel_launch(void* const* d_in, const int* in_sizes, int n_in,
                              void* d_out, int out_size, void* d_ws,
                              size_t ws_size, hipStream_t stream) {
  (void)in_sizes; (void)n_in; (void)d_ws; (void)ws_size; (void)out_size;
  const float* z  = (const float*)d_in[0];
  const float* WQ = (const float*)d_in[1];
  const float* WK = (const float*)d_in[2];
  float* out = (float*)d_out;

  const int nProblems = 4096;          // NH * b * h  (= NH * b * w)
  const size_t shBytes = 61440;        // 60 KB dynamic LDS
  axial_mha_kernel<<<nProblems, 256, shBytes, stream>>>(z, WQ, WK, out);
}
